// W8DXLinear_18502719111672
// MI455X (gfx1250) — 8-bit matrix operand order repaired and checked by reading, compiled, not run on hardware
//
#include <hip/hip_runtime.h>
#include <hip/hip_bf16.h>

typedef int v8i __attribute__((ext_vector_type(8)));

#define N_ROWS  4096
#define IN_DIM  4096
#define OUT_DIM 4096
#define KOUT    16

__global__ __launch_bounds__(256) void zero_u32_kernel(unsigned* __restrict__ p, int n) {
    int i = blockIdx.x * 256 + threadIdx.x;
    if (i < n) p[i] = 0u;
}

__global__ __launch_bounds__(256) void colmax_kernel(const float* __restrict__ x,
                                                     unsigned* __restrict__ colmax) {
    int c  = blockIdx.x * 256 + threadIdx.x;
    int r0 = blockIdx.y * 128;
    float m = 0.0f;
    for (int r = 0; r < 128; ++r)
        m = fmaxf(m, fabsf(x[(size_t)(r0 + r) * IN_DIM + c]));
    atomicMax(&colmax[c], __float_as_uint(m));
}

__global__ __launch_bounds__(256) void topk_kernel(const unsigned* __restrict__ colmax,
                                                   int* __restrict__ idx_out,
                                                   int* __restrict__ mask) {
    __shared__ float cm[IN_DIM];
    __shared__ float sv[256];
    __shared__ int   si[256];
    int t = threadIdx.x;
    for (int i = 0; i < 16; ++i) {
        int c = i * 256 + t;
        cm[c] = __uint_as_float(colmax[c]);
        mask[c] = 0;
    }
    __syncthreads();
    for (int round = 0; round < KOUT; ++round) {
        float best = -1.0f; int bi = 0x7fffffff;
        for (int i = 0; i < 16; ++i) {
            int c = i * 256 + t;
            float v = cm[c];
            if (v > best) { best = v; bi = c; }
        }
        sv[t] = best; si[t] = bi;
        __syncthreads();
        for (int s = 128; s > 0; s >>= 1) {
            if (t < s) {
                if (sv[t + s] > sv[t] || (sv[t + s] == sv[t] && si[t + s] < si[t])) {
                    sv[t] = sv[t + s]; si[t] = si[t + s];
                }
            }
            __syncthreads();
        }
        int win = si[0];
        if (t == 0) {
            idx_out[round] = win;
            mask[win] = 1;
            cm[win] = -1.0f;
        }
        __syncthreads();
    }
}

__global__ __launch_bounds__(256) void quantize_kernel(const float* __restrict__ x,
                                                       const int* __restrict__ mask,
                                                       const int* __restrict__ idx,
                                                       signed char* __restrict__ xq,
                                                       float* __restrict__ rowmax,
                                                       float* __restrict__ xu) {
    int m = blockIdx.x;
    int t = threadIdx.x;
    const float* xr = x + (size_t)m * IN_DIM;
    __shared__ float red[256];
    float mx = 0.0f;
    for (int i = 0; i < 16; ++i) {
        int c = i * 256 + t;
        if (!mask[c]) mx = fmaxf(mx, fabsf(xr[c]));
    }
    red[t] = mx;
    __syncthreads();
    for (int s = 128; s > 0; s >>= 1) {
        if (t < s) red[t] = fmaxf(red[t], red[t + s]);
        __syncthreads();
    }
    float rmx   = red[0];
    float scale = 127.0f / fmaxf(rmx, 1e-30f);
    for (int i = 0; i < 16; ++i) {
        int c = i * 256 + t;
        float q = mask[c] ? 0.0f : xr[c] * scale;
        xq[(size_t)m * IN_DIM + c] = (signed char)(int)q;
    }
    if (t < KOUT) xu[m * KOUT + t] = xr[idx[t]];
    if (t == 0)   rowmax[m] = rmx;
}

__global__ __launch_bounds__(256) void transpose_kernel(const signed char* __restrict__ w,
                                                        signed char* __restrict__ wT) {
    __shared__ signed char tile[64][65];
    int k0 = blockIdx.y * 64, n0 = blockIdx.x * 64;
    int t = threadIdx.x;
    for (int i = 0; i < 16; ++i) {
        int idx = i * 256 + t;
        int k = idx >> 6, n = idx & 63;
        tile[k][n] = w[(size_t)(k0 + k) * OUT_DIM + n0 + n];
    }
    __syncthreads();
    for (int i = 0; i < 16; ++i) {
        int idx = i * 256 + t;
        int n = idx >> 6, k = idx & 63;
        wT[(size_t)(n0 + n) * IN_DIM + k0 + k] = tile[k][n];
    }
}

__global__ __launch_bounds__(256) void wu_kernel(const signed char* __restrict__ w,
                                                 const float* __restrict__ wmax,
                                                 const int* __restrict__ idx,
                                                 float* __restrict__ wu) {
    int n = blockIdx.x * 256 + threadIdx.x;
    float s = wmax[n] * (1.0f / 127.0f);
    for (int j = 0; j < KOUT; ++j)
        wu[j * OUT_DIM + n] = (float)w[(size_t)idx[j] * OUT_DIM + n] * s;
}

__global__ __launch_bounds__(256) void gemm_kernel(const signed char* __restrict__ xq,
                                                   const signed char* __restrict__ wT,
                                                   const float* __restrict__ rowmax,
                                                   const float* __restrict__ wmax,
                                                   const float* __restrict__ bias,
                                                   const float* __restrict__ xu,
                                                   const float* __restrict__ wu,
                                                   float* __restrict__ out) {
    __shared__ signed char Asl[256 * 64];
    __shared__ signed char Bsl[128 * 64];
    __shared__ float xus[256 * KOUT];
    __shared__ float wus[KOUT * 128];

    const int t    = threadIdx.x;
    const int lane = t & 31;
    const int wid  = t >> 5;
    const int wm   = (wid & 3) * 64;
    const int wn   = (wid >> 2) * 64;
    const int Mblk = blockIdx.y * 256;
    const int Nblk = blockIdx.x * 128;

    for (int i = 0; i < 16; ++i) {
        int idx = i * 256 + t;
        xus[idx] = xu[Mblk * KOUT + idx];
    }
    for (int i = 0; i < 8; ++i) {
        int idx = i * 256 + t;
        int j = idx >> 7, nl = idx & 127;
        wus[idx] = wu[j * OUT_DIM + Nblk + nl];
    }

    v8i acc[4][4];
    const v8i vzero = {0, 0, 0, 0, 0, 0, 0, 0};
#pragma unroll
    for (int i = 0; i < 4; ++i)
#pragma unroll
        for (int j = 0; j < 4; ++j) acc[i][j] = vzero;

    const int mrow   = lane & 15;
    const int khalfA = (lane >> 4) * 8;
    const int khalfB = (lane >> 4) * 8;

    for (int k0 = 0; k0 < IN_DIM; k0 += 64) {
        __syncthreads();
#pragma unroll
        for (int i = 0; i < 4; ++i) {
            int idx = i * 256 + t;
            int row = idx >> 2, ch = idx & 3;
            *(int4*)(&Asl[row * 64 + ch * 16]) =
                *(const int4*)(xq + (size_t)(Mblk + row) * IN_DIM + k0 + ch * 16);
        }
#pragma unroll
        for (int i = 0; i < 2; ++i) {
            int idx = i * 256 + t;
            int row = idx >> 2, ch = idx & 3;
            *(int4*)(&Bsl[row * 64 + ch * 16]) =
                *(const int4*)(wT + (size_t)(Nblk + row) * IN_DIM + k0 + ch * 16);
        }
        __syncthreads();

        v8i a[4], b[4];
#pragma unroll
        for (int tm = 0; tm < 4; ++tm) {
            const int* p = (const int*)(&Asl[(wm + tm * 16 + mrow) * 64 + khalfA]);
            a[tm][0] = p[0];  a[tm][1] = p[1];
            a[tm][2] = p[4];  a[tm][3] = p[5];
            a[tm][4] = p[8];  a[tm][5] = p[9];
            a[tm][6] = p[12]; a[tm][7] = p[13];
        }
#pragma unroll
        for (int tn = 0; tn < 4; ++tn) {
            const int* p = (const int*)(&Bsl[(wn + tn * 16 + mrow) * 64 + khalfB]);
            b[tn][0] = p[0];  b[tn][1] = p[1];  b[tn][2] = p[4];  b[tn][3] = p[5];
            b[tn][4] = p[8];  b[tn][5] = p[9];  b[tn][6] = p[12]; b[tn][7] = p[13];
        }
#pragma unroll
        for (int tm = 0; tm < 4; ++tm)
#pragma unroll
            for (int tn = 0; tn < 4; ++tn)
                acc[tm][tn] = __builtin_amdgcn_wmma_i32_16x16x64_iu8(
                    true, a[tm], true, b[tn], acc[tm][tn], false, false);
    }

    const float inv = 1.0f / (127.0f * 127.0f);
#pragma unroll
    for (int tn = 0; tn < 4; ++tn) {
        int nl = wn + tn * 16 + (lane & 15);
        int n  = Nblk + nl;
        float wms = wmax[n] * inv;
        float bs  = bias[n];
#pragma unroll
        for (int tm = 0; tm < 4; ++tm) {
#pragma unroll
            for (int r = 0; r < 8; ++r) {
                int ml = wm + tm * 16 + r + ((lane >> 4) << 3);
                int m  = Mblk + ml;
                float s = (float)acc[tm][tn][r] * rowmax[m] * wms;
                float u = 0.0f;
#pragma unroll
                for (int j = 0; j < KOUT; ++j)
                    u += xus[ml * KOUT + j] * wus[j * 128 + nl];
                out[(size_t)m * OUT_DIM + n] = s + u + bs;
            }
        }
    }
}

extern "C" void kernel_launch(void* const* d_in, const int* in_sizes, int n_in,
                              void* d_out, int out_size, void* d_ws, size_t ws_size,
                              hipStream_t stream) {
    const float*       x    = (const float*)d_in[0];
    const signed char* wq   = (const signed char*)d_in[1];
    const float*       wmax = (const float*)d_in[2];
    const float*       bias = (const float*)d_in[3];
    float*             out  = (float*)d_out;

    char* ws = (char*)d_ws;
    size_t off = 0;
    unsigned*    colmax = (unsigned*)(ws + off);    off += (size_t)IN_DIM * 4;
    int*         idx    = (int*)(ws + off);         off += 256;
    int*         mask   = (int*)(ws + off);         off += (size_t)IN_DIM * 4;
    float*       rowmax = (float*)(ws + off);       off += (size_t)N_ROWS * 4;
    float*       xu     = (float*)(ws + off);       off += (size_t)N_ROWS * KOUT * 4;
    float*       wu     = (float*)(ws + off);       off += (size_t)KOUT * OUT_DIM * 4;
    signed char* xq     = (signed char*)(ws + off); off += (size_t)N_ROWS * IN_DIM;
    signed char* wT     = (signed char*)(ws + off); off += (size_t)OUT_DIM * IN_DIM;

    zero_u32_kernel<<<IN_DIM / 256, 256, 0, stream>>>(colmax, IN_DIM);
    colmax_kernel<<<dim3(IN_DIM / 256, N_ROWS / 128), 256, 0, stream>>>(x, colmax);
    topk_kernel<<<1, 256, 0, stream>>>(colmax, idx, mask);
    quantize_kernel<<<N_ROWS, 256, 0, stream>>>(x, mask, idx, xq, rowmax, xu);
    transpose_kernel<<<dim3(OUT_DIM / 64, IN_DIM / 64), 256, 0, stream>>>(wq, wT);
    wu_kernel<<<OUT_DIM / 256, 256, 0, stream>>>(wq, wmax, idx, wu);
    gemm_kernel<<<dim3(OUT_DIM / 128, N_ROWS / 256), 256, 0, stream>>>(
        xq, wT, rowmax, wmax, bias, xu, wu, out);
}
